// Att_BiLSTM_CRF_38001870635616
// MI455X (gfx1250) — compile-verified, matrix operand order checked by reading, not run on hardware
//
#include <hip/hip_runtime.h>
#include <hip/hip_bf16.h>
#include <math.h>

// ---------------------------------------------------------------------------
// Att-BiLSTM-CRF for MI455X (gfx1250, wave32, WMMA + TDM).
// Whole working set (~50MB) is L2-resident (192MB), so all GEMM operands are
// kept in bf16 fragment-ready layouts: every WMMA fragment = two 16B vector
// loads (global_load_b128 / ds_load_b128) + one shuffle. W tiles are staged
// to LDS by the Tensor Data Mover (tensor_load_to_lds + s_wait_tensorcnt).
// ---------------------------------------------------------------------------

typedef __attribute__((ext_vector_type(16))) __bf16 v16bf;
typedef __attribute__((ext_vector_type(8)))  __bf16 v8bf;
typedef __attribute__((ext_vector_type(8)))  float  v8f;
typedef __attribute__((ext_vector_type(4)))  unsigned int u32x4;
typedef __attribute__((ext_vector_type(8)))  int i32x8;
typedef __attribute__((ext_vector_type(4)))  int i32x4;

#if defined(__has_builtin)
#if __has_builtin(__builtin_amdgcn_tensor_load_to_lds)
#define USE_TDM 1
#endif
#endif

#define Bv 32
#define Sv 512
#define Ev 256
#define Hv 256
#define Tv 9
#define BSv (Bv * Sv)          // 16384
#define G4H (4 * Hv)           // 1024
#define ATTD 128

#define BCAT(lo, hi) __builtin_shufflevector(lo, hi, 0, 1, 2, 3, 4, 5, 6, 7, \
                                             8, 9, 10, 11, 12, 13, 14, 15)

__device__ __forceinline__ v8f wmma_bf16(v16bf a, v16bf b, v8f c) {
  return __builtin_amdgcn_wmma_f32_16x16x32_bf16(false, a, false, b,
                                                 (short)0, c, false, false);
}

__device__ __forceinline__ float sigf(float x) { return 1.0f / (1.0f + expf(-x)); }

// ---------------------------------------------------------------------------
// Embedding gather -> bf16 activations
// ---------------------------------------------------------------------------
__global__ void embed_kernel(const int* __restrict__ tok,
                             const float* __restrict__ emb,
                             __bf16* __restrict__ x) {
  int i = blockIdx.x * blockDim.x + threadIdx.x;
  if (i < BSv * Ev) {
    int bs = i >> 8, d = i & 255;
    x[i] = (__bf16)emb[(size_t)tok[bs] * Ev + d];
  }
}

__global__ void f2bf_kernel(const float* __restrict__ in,
                            __bf16* __restrict__ out, int n) {
  int i = blockIdx.x * blockDim.x + threadIdx.x;
  if (i < n) out[i] = (__bf16)in[i];
}

__global__ void bias2_kernel(const float* __restrict__ a,
                             const float* __restrict__ b,
                             float* __restrict__ o, int n) {
  int i = blockIdx.x * blockDim.x + threadIdx.x;
  if (i < n) o[i] = a[i] + b[i];
}

// ---------------------------------------------------------------------------
// WMMA GEMM: C[M,N] = A[M,K] @ W[N,K]^T + bias[N], all operands bf16.
// Block tile 64x32 (8 waves, 4m x 2n of 16x16 WMMA tiles), K step 32.
// A fragments: direct b128 vector loads from L2 (row-major, contiguous in K).
// W tile (32n x 32k, 2KB): DMA'd into LDS by the Tensor Data Mover each
// K-step (wave 0 issues, s_wait_tensorcnt, barrier), consumed as b128 ds
// loads (row-major => contiguous in K).
// flags: 1 = write f32 C, 2 = write bf16 C, 4 = write bf16 C transposed
// per-batch ([b][n][s], for the attention V operand).
// ---------------------------------------------------------------------------
__global__ __launch_bounds__(256) void gemm_kernel(
    const __bf16* __restrict__ Ab, const __bf16* __restrict__ Wb,
    const float* __restrict__ bias, float* __restrict__ Cf,
    __bf16* __restrict__ Cb, int flags, int M, int N, int K) {
  __shared__ __bf16 Ws[32][32];
  const int tid = threadIdx.x;
  const int lane = tid & 31;
  const int w = tid >> 5;
  const int wm = w & 3, wn = w >> 2;
  const int m0 = blockIdx.y * 64;
  const int n0 = blockIdx.x * 32;
  const int hl = lane >> 4;       // lane half
  const int kg = hl * 8;
  const int ml = lane & 15;
  const __bf16* arow = Ab + (size_t)(m0 + wm * 16 + ml) * K;
  v8f acc = {};
#ifdef USE_TDM
  // D# group1 (ISA 08 s8.4): data_size=2B, tensor K x N, tile 32x32, stride K.
  i32x8 g1 = {};
  g1[0] = (int)(1u << 16);                                   // data_size=1 (2B)
  g1[1] = (int)(((unsigned)K & 0xFFFFu) << 16);              // tensor_dim0 lo
  g1[2] = (int)((((unsigned)K >> 16) & 0xFFFFu) |
                (((unsigned)N & 0xFFFFu) << 16));            // dim0 hi | dim1 lo
  g1[3] = (int)((((unsigned)N >> 16) & 0xFFFFu) | (32u << 16)); // dim1 hi|tile0
  g1[4] = 32;                                                // tile_dim1=32
  g1[5] = K;                                                 // dim0 stride lo
  const unsigned int ldsa = (unsigned int)(size_t)(&Ws[0][0]);
  const i32x4 z4 = {};
  const i32x8 z8 = {};
#endif
  for (int k0 = 0; k0 < K; k0 += 32) {
#ifdef USE_TDM
    if (w == 0) {
      unsigned long long ga =
          (unsigned long long)(size_t)(Wb + (size_t)n0 * K + k0);
      u32x4 g0;
      g0[0] = 1u;                                            // count=1
      g0[1] = ldsa;                                          // lds_addr
      g0[2] = (unsigned int)ga;                              // global_addr lo
      g0[3] = (unsigned int)((ga >> 32) & 0x1FFFFFFu) | (2u << 30); // hi|type=2
      __builtin_amdgcn_tensor_load_to_lds(g0, g1, z4, z4, z8, 0);
      __builtin_amdgcn_s_wait_tensorcnt(0);
    }
#else
    for (int e = tid; e < 32 * 32; e += 256) {
      int r = e >> 5, kk = e & 31;
      Ws[r][kk] = Wb[(size_t)(n0 + r) * K + k0 + kk];
    }
#endif
    __syncthreads();
    __builtin_prefetch(arow + k0 + 64, 0, 1);  // global_prefetch next slabs
    v8bf alo = *(const v8bf*)(arow + k0 + kg);
    v8bf ahi = *(const v8bf*)(arow + k0 + 16 + kg);
    v8bf blo = *(const v8bf*)&Ws[wn * 16 + ml][kg];
    v8bf bhi = *(const v8bf*)&Ws[wn * 16 + ml][16 + kg];
    acc = wmma_bf16(BCAT(alo, ahi), BCAT(blo, bhi), acc);
    __syncthreads();
  }
  const int col = n0 + wn * 16 + ml;
  const float bcol = bias[col];
#pragma unroll
  for (int r = 0; r < 8; ++r) {
    int row = m0 + wm * 16 + r + hl * 8;
    float v = acc[r] + bcol;
    if (flags & 1) Cf[(size_t)row * N + col] = v;
    if (flags & 2) Cb[(size_t)row * N + col] = (__bf16)v;
    if (flags & 4) {
      int bi = row >> 9, s = row & 511;
      Cb[((size_t)bi * N + col) * Sv + s] = (__bf16)v;
    }
  }
}

// ---------------------------------------------------------------------------
// Persistent bidirectional LSTM scan, grid=2 (dir), block=1024 (32 waves).
// h state lives in LDS in WMMA-A fragment-ready bf16 layout: phase-1 A
// fragments are two ds_load_b128 per wave, Whh B fragments are two
// global_load_b128 from the bf16 weight image in L2.
// ---------------------------------------------------------------------------
__global__ __launch_bounds__(1024) void lstm_scan_kernel(
    const float* __restrict__ gx0, const float* __restrict__ gx1,
    const __bf16* __restrict__ whh0, const __bf16* __restrict__ whh1,
    __bf16* __restrict__ out, float* __restrict__ cbase,
    float* __restrict__ gbase) {
  const int dir = blockIdx.x;
  const float* gx = dir ? gx1 : gx0;
  const __bf16* whh = dir ? whh1 : whh0;
  float* cst = cbase + (size_t)dir * Bv * Hv;
  float* gst = gbase + (size_t)dir * Bv * G4H;
  // h in fragment layout: [mtile][kchunk][lane][16]
  __shared__ __bf16 hfrag[2][8][32][16];
  const int tid = threadIdx.x;
  const int lane = tid & 31;
  const int w = tid >> 5;
  const int hl = lane >> 4;
  const int kg = hl * 8;
  const int ml = lane & 15;
  for (int e = tid; e < 2 * 8 * 32 * 16; e += 1024)
    ((__bf16*)hfrag)[e] = (__bf16)0.0f;
  for (int e = tid; e < Bv * Hv; e += 1024) cst[e] = 0.0f;
  __syncthreads();
  for (int t = 0; t < Sv; ++t) {
    const int ts = dir ? (Sv - 1 - t) : t;
    // phase 1: g = h @ Whh^T + gx  (128 tiles, 4 per wave, 8 WMMAs each)
#pragma unroll
    for (int j = 0; j < 4; ++j) {
      const int tile = w + 32 * j;
      const int mtile = tile & 1;
      const int ntile = tile >> 1;
      const __bf16* wrow = whh + (size_t)(ntile * 16 + ml) * Hv;
      v8f acc = {};
#pragma unroll
      for (int kc = 0; kc < 8; ++kc) {
        const int k0 = kc * 32;
        v8bf alo = *(const v8bf*)&hfrag[mtile][kc][lane][0];
        v8bf ahi = *(const v8bf*)&hfrag[mtile][kc][lane][8];
        v8bf blo = *(const v8bf*)(wrow + k0 + kg);
        v8bf bhi = *(const v8bf*)(wrow + k0 + 16 + kg);
        acc = wmma_bf16(BCAT(alo, ahi), BCAT(blo, bhi), acc);
      }
#pragma unroll
      for (int r = 0; r < 8; ++r) {
        int bi = mtile * 16 + r + hl * 8;
        int nn = ntile * 16 + ml;
        gst[bi * G4H + nn] = acc[r] + gx[((size_t)bi * Sv + ts) * G4H + nn];
      }
    }
    __syncthreads();
    // phase 2: gates, state update, h -> fragment LDS + bf16 output
    for (int e = tid; e < Bv * Hv; e += 1024) {
      int bi = e >> 8, hh = e & 255;
      float gi = gst[bi * G4H + hh];
      float gf = gst[bi * G4H + 256 + hh];
      float gg = gst[bi * G4H + 512 + hh];
      float go = gst[bi * G4H + 768 + hh];
      float co = cst[e];
      float cn = sigf(gf) * co + sigf(gi) * tanhf(gg);
      float hn = sigf(go) * tanhf(cn);
      cst[e] = cn;
      int m = bi & 15, mt = bi >> 4;
      int kc = hh >> 5, kk = hh & 31;
      int hl2 = (kk >> 3) & 1;
      int pos = (kk & 7) | ((kk >> 4) << 3);
      hfrag[mt][kc][m + hl2 * 16][pos] = (__bf16)hn;
      out[((size_t)bi * Sv + ts) * (2 * Hv) + dir * Hv + hh] = (__bf16)hn;
    }
    __syncthreads();
  }
}

// ---------------------------------------------------------------------------
// Attention: per (batch, 16-query tile) block. QK^T and P@V on WMMA; Q/K/V
// fragments are direct b128 vector loads (V pre-transposed per batch), P is
// built row-major in LDS so its fragments are two ds_load_b128.
// ---------------------------------------------------------------------------
__global__ __launch_bounds__(256) void attn_kernel(
    const __bf16* __restrict__ Qb, const __bf16* __restrict__ Kb,
    const __bf16* __restrict__ Vt, const int* __restrict__ mask,
    float* __restrict__ ctx) {
  const int qt = blockIdx.x;   // 0..31
  const int bi = blockIdx.y;   // 0..31
  __shared__ float Sc[16][520];
  __shared__ __bf16 Ps[16][520];
  const int tid = threadIdx.x;
  const int lane = tid & 31;
  const int w = tid >> 5;
  const int hl = lane >> 4, kg = hl * 8, ml = lane & 15;
  const float scale = 0.0883883476483184f;  // 1/sqrt(128)
  const __bf16* qrow = Qb + ((size_t)(bi * Sv) + qt * 16 + ml) * ATTD;
  for (int kt = w; kt < 32; kt += 8) {
    const int key = kt * 16 + ml;
    const __bf16* krow = Kb + ((size_t)(bi * Sv) + key) * ATTD;
    v8f acc = {};
#pragma unroll
    for (int k0 = 0; k0 < ATTD; k0 += 32) {
      v8bf alo = *(const v8bf*)(qrow + k0 + kg);
      v8bf ahi = *(const v8bf*)(qrow + k0 + 16 + kg);
      v8bf blo = *(const v8bf*)(krow + k0 + kg);
      v8bf bhi = *(const v8bf*)(krow + k0 + 16 + kg);
      acc = wmma_bf16(BCAT(alo, ahi), BCAT(blo, bhi), acc);
    }
    int mk = mask[bi * Sv + key];
#pragma unroll
    for (int r = 0; r < 8; ++r)
      Sc[r + hl * 8][key] = (mk > 0) ? acc[r] * scale : -1e9f;
  }
  __syncthreads();
  if (tid < 16) {
    float mx = -1e30f;
    for (int k = 0; k < Sv; ++k) mx = fmaxf(mx, Sc[tid][k]);
    float sm = 0.0f;
    for (int k = 0; k < Sv; ++k) {
      float p = expf(Sc[tid][k] - mx);
      Sc[tid][k] = p;
      sm += p;
    }
    float inv = 1.0f / sm;
    for (int k = 0; k < Sv; ++k) Ps[tid][k] = (__bf16)(Sc[tid][k] * inv);
  }
  __syncthreads();
  {
    const int nt = w;  // 8 waves x 16 cols = ATTD
    const __bf16* vrow = Vt + ((size_t)bi * ATTD + nt * 16 + ml) * Sv;
    v8f acc = {};
    for (int k0 = 0; k0 < Sv; k0 += 32) {
      v8bf alo = *(const v8bf*)&Ps[ml][k0 + kg];
      v8bf ahi = *(const v8bf*)&Ps[ml][k0 + 16 + kg];
      v8bf blo = *(const v8bf*)(vrow + k0 + kg);
      v8bf bhi = *(const v8bf*)(vrow + k0 + 16 + kg);
      acc = wmma_bf16(BCAT(alo, ahi), BCAT(blo, bhi), acc);
    }
#pragma unroll
    for (int r = 0; r < 8; ++r)
      ctx[((size_t)(bi * Sv) + qt * 16 + r + hl * 8) * ATTD + nt * 16 + ml] =
          acc[r];
  }
}

// ---------------------------------------------------------------------------
// Emission head (T=9): scalar dot products.
// ---------------------------------------------------------------------------
__global__ void emis_kernel(const float* __restrict__ ctx,
                            const float* __restrict__ Wc,
                            const float* __restrict__ bc,
                            float* __restrict__ em) {
  int i = blockIdx.x * blockDim.x + threadIdx.x;
  if (i < BSv * Tv) {
    int bs = i / Tv, t = i % Tv;
    float acc = 0.0f;
    for (int k = 0; k < ATTD; ++k) acc += ctx[(size_t)bs * ATTD + k] * Wc[t * ATTD + k];
    em[(size_t)bs * Tv + t] = acc + bc[t];
  }
}

// ---------------------------------------------------------------------------
// CRF forward (logZ + numerator) and Viterbi, one wave per batch item.
// ---------------------------------------------------------------------------
__global__ __launch_bounds__(32) void crf_kernel(
    const float* __restrict__ em, const int* __restrict__ labels,
    const int* __restrict__ mask, const float* __restrict__ cs,
    const float* __restrict__ ce, const float* __restrict__ ctr,
    float* __restrict__ out_tokens, float* __restrict__ llh) {
  const int b = blockIdx.x;
  const int t = threadIdx.x;
  __shared__ float trs[Tv][Tv];
  __shared__ float sbuf[32];
  __shared__ float vbuf[32];
  __shared__ unsigned char bp[Sv][Tv];
  for (int e = t; e < Tv * Tv; e += 32) trs[e / Tv][e % Tv] = ctr[e];
  __syncthreads();
  const float* emb = em + (size_t)b * Sv * Tv;
  const int* mkb = mask + b * Sv;
  const int tc = (t < Tv) ? t : 0;
  float alpha = (t < Tv) ? cs[t] + emb[t] : -1e30f;
  float valp = alpha;
  for (int s = 1; s < Sv; ++s) {
    sbuf[t] = alpha;
    vbuf[t] = valp;
    __syncthreads();
    float emt = (t < Tv) ? emb[s * Tv + t] : 0.0f;
    float mt = (float)mkb[s];
    float mx = -1e30f, vmx = -1e30f;
    int varg = 0;
    for (int j = 0; j < Tv; ++j) {
      float c1 = sbuf[j] + trs[j][tc];
      mx = fmaxf(mx, c1);
      if (c1 > vmx) { vmx = c1; varg = j; }
    }
    float sm = 0.0f;
    for (int j = 0; j < Tv; ++j) sm += expf(sbuf[j] + trs[j][tc] - mx);
    float anew = mx + logf(sm) + emt;
    float vnew = vmx + emt;
    if (t < Tv) {
      alpha = (mt > 0.0f) ? anew : alpha;
      bp[s][t] = (unsigned char)((mt > 0.0f) ? varg : t);
      valp = (mt > 0.0f) ? vnew : valp;
    }
    __syncthreads();
  }
  sbuf[t] = alpha;
  vbuf[t] = valp;
  __syncthreads();
  if (t == 0) {
    float mx = -1e30f;
    for (int j = 0; j < Tv; ++j) mx = fmaxf(mx, sbuf[j] + ce[j]);
    float sm = 0.0f;
    for (int j = 0; j < Tv; ++j) sm += expf(sbuf[j] + ce[j] - mx);
    float logz = mx + logf(sm);
    const int* lb = labels + b * Sv;
    float num = cs[lb[0]] + emb[lb[0]];
    int len = 0;
    for (int s = 0; s < Sv; ++s) len += (mkb[s] != 0);
    for (int s = 1; s < Sv; ++s) {
      float mt = (float)mkb[s];
      num += (trs[lb[s - 1]][lb[s]] + emb[s * Tv + lb[s]]) * mt;
    }
    num += ce[lb[len - 1]];
    llh[b] = num - logz;
    float bmx = -1e30f;
    int last = 0;
    for (int j = 0; j < Tv; ++j) {
      float v2 = vbuf[j] + ce[j];
      if (v2 > bmx) { bmx = v2; last = j; }
    }
    int tag = last;
    out_tokens[b * Sv + (Sv - 1)] = (float)(tag * (mkb[Sv - 1] != 0 ? 1 : 0));
    for (int s = Sv - 2; s >= 0; --s) {
      tag = bp[s + 1][tag];
      out_tokens[b * Sv + s] = (float)(tag * (mkb[s] != 0 ? 1 : 0));
    }
  }
}

__global__ void loss_kernel(const float* __restrict__ llh, float* __restrict__ out) {
  __shared__ float sb[32];
  sb[threadIdx.x] = llh[threadIdx.x];
  __syncthreads();
  if (threadIdx.x == 0) {
    float s = 0.0f;
    for (int i = 0; i < Bv; ++i) s += sb[i];
    out[0] = -s / (float)Bv;
  }
}

// ---------------------------------------------------------------------------
extern "C" void kernel_launch(void* const* d_in, const int* in_sizes, int n_in,
                              void* d_out, int out_size, void* d_ws, size_t ws_size,
                              hipStream_t stream) {
  const int* tok    = (const int*)d_in[0];
  const int* labels = (const int*)d_in[1];
  const int* amask  = (const int*)d_in[2];
  const float* emb  = (const float*)d_in[3];
  const float* Wih[2][2] = {{(const float*)d_in[4],  (const float*)d_in[8]},
                            {(const float*)d_in[12], (const float*)d_in[16]}};
  const float* Whh[2][2] = {{(const float*)d_in[5],  (const float*)d_in[9]},
                            {(const float*)d_in[13], (const float*)d_in[17]}};
  const float* bih[2][2] = {{(const float*)d_in[6],  (const float*)d_in[10]},
                            {(const float*)d_in[14], (const float*)d_in[18]}};
  const float* bhh[2][2] = {{(const float*)d_in[7],  (const float*)d_in[11]},
                            {(const float*)d_in[15], (const float*)d_in[19]}};
  const float* Wq = (const float*)d_in[20];
  const float* bq = (const float*)d_in[21];
  const float* Wk = (const float*)d_in[22];
  const float* bk = (const float*)d_in[23];
  const float* Wv = (const float*)d_in[24];
  const float* bv = (const float*)d_in[25];
  const float* Wc = (const float*)d_in[26];
  const float* bc = (const float*)d_in[27];
  const float* cs = (const float*)d_in[28];
  const float* ce = (const float*)d_in[29];
  const float* ctr = (const float*)d_in[30];

  float* ws = (float*)d_ws;
  size_t off = 0;
  float* gxf = ws + off; off += (size_t)BSv * G4H;       // fwd input proj (f32)
  float* gxb = ws + off; off += (size_t)BSv * G4H;       // bwd input proj (f32)
  float* ctx = ws + off; off += (size_t)BSv * ATTD;
  float* em  = ws + off; off += (size_t)BSv * Tv;
  float* llh = ws + off; off += 64;
  float* cws = ws + off; off += (size_t)2 * Bv * Hv;     // LSTM cell state
  float* gws = ws + off; off += (size_t)2 * Bv * G4H;    // per-step gates
  float* biasA = ws + off; off += G4H;
  float* biasB = ws + off; off += G4H;
  __bf16* x0b  = (__bf16*)(ws + off); off += (size_t)BSv * Ev / 2;
  __bf16* x1b  = (__bf16*)(ws + off); off += (size_t)BSv * 2 * Hv / 2;
  __bf16* x2b  = (__bf16*)(ws + off); off += (size_t)BSv * 2 * Hv / 2;
  __bf16* qbf  = (__bf16*)(ws + off); off += (size_t)BSv * ATTD / 2;
  __bf16* kbf  = (__bf16*)(ws + off); off += (size_t)BSv * ATTD / 2;
  __bf16* vtbf = (__bf16*)(ws + off); off += (size_t)BSv * ATTD / 2;
  __bf16* wihA = (__bf16*)(ws + off); off += (size_t)G4H * 512 / 2;
  __bf16* wihB = (__bf16*)(ws + off); off += (size_t)G4H * 512 / 2;
  __bf16* whhA = (__bf16*)(ws + off); off += (size_t)G4H * Hv / 2;
  __bf16* whhB = (__bf16*)(ws + off); off += (size_t)G4H * Hv / 2;
  __bf16* wqb  = (__bf16*)(ws + off); off += (size_t)ATTD * 512 / 2;
  __bf16* wkb  = (__bf16*)(ws + off); off += (size_t)ATTD * 512 / 2;
  __bf16* wvb  = (__bf16*)(ws + off); off += (size_t)ATTD * 512 / 2;

  const int WHH_N = G4H * Hv;

  // 1. embedding gather -> bf16
  embed_kernel<<<(BSv * Ev + 255) / 256, 256, 0, stream>>>(tok, emb, x0b);

  for (int l = 0; l < 2; ++l) {
    const int Kin = (l == 0) ? Ev : 2 * Hv;
    const __bf16* xin = (l == 0) ? x0b : x1b;
    __bf16* xout = (l == 0) ? x1b : x2b;
    // weights/bias prep
    f2bf_kernel<<<(G4H * Kin + 255) / 256, 256, 0, stream>>>(Wih[l][0], wihA, G4H * Kin);
    f2bf_kernel<<<(G4H * Kin + 255) / 256, 256, 0, stream>>>(Wih[l][1], wihB, G4H * Kin);
    f2bf_kernel<<<(WHH_N + 255) / 256, 256, 0, stream>>>(Whh[l][0], whhA, WHH_N);
    f2bf_kernel<<<(WHH_N + 255) / 256, 256, 0, stream>>>(Whh[l][1], whhB, WHH_N);
    bias2_kernel<<<(G4H + 255) / 256, 256, 0, stream>>>(bih[l][0], bhh[l][0], biasA, G4H);
    bias2_kernel<<<(G4H + 255) / 256, 256, 0, stream>>>(bih[l][1], bhh[l][1], biasB, G4H);
    // input projections (WMMA + TDM)
    gemm_kernel<<<dim3(G4H / 32, BSv / 64), 256, 0, stream>>>(
        xin, wihA, biasA, gxf, nullptr, 1, BSv, G4H, Kin);
    gemm_kernel<<<dim3(G4H / 32, BSv / 64), 256, 0, stream>>>(
        xin, wihB, biasB, gxb, nullptr, 1, BSv, G4H, Kin);
    // bidirectional recurrent scan
    lstm_scan_kernel<<<2, 1024, 0, stream>>>(gxf, gxb, whhA, whhB, xout, cws, gws);
  }

  // q/k/v projections (K=512, N=128) -> bf16 (V transposed per batch)
  f2bf_kernel<<<(ATTD * 512 + 255) / 256, 256, 0, stream>>>(Wq, wqb, ATTD * 512);
  f2bf_kernel<<<(ATTD * 512 + 255) / 256, 256, 0, stream>>>(Wk, wkb, ATTD * 512);
  f2bf_kernel<<<(ATTD * 512 + 255) / 256, 256, 0, stream>>>(Wv, wvb, ATTD * 512);
  gemm_kernel<<<dim3(ATTD / 32, BSv / 64), 256, 0, stream>>>(
      x2b, wqb, bq, nullptr, qbf, 2, BSv, ATTD, 2 * Hv);
  gemm_kernel<<<dim3(ATTD / 32, BSv / 64), 256, 0, stream>>>(
      x2b, wkb, bk, nullptr, kbf, 2, BSv, ATTD, 2 * Hv);
  gemm_kernel<<<dim3(ATTD / 32, BSv / 64), 256, 0, stream>>>(
      x2b, wvb, bv, nullptr, vtbf, 4, BSv, ATTD, 2 * Hv);

  // attention
  attn_kernel<<<dim3(Sv / 16, Bv), 256, 0, stream>>>(qbf, kbf, vtbf, amask, ctx);

  // emissions
  emis_kernel<<<(BSv * Tv + 255) / 256, 256, 0, stream>>>(ctx, Wc, bc, em);

  // CRF loss + Viterbi decode
  float* out_f = (float*)d_out;
  crf_kernel<<<Bv, 32, 0, stream>>>(em, labels, amask, cs, ce, ctr, out_f + 1, llh);
  loss_kernel<<<1, 32, 0, stream>>>(llh, out_f);
}
